// BSplineKAN_44719199486017
// MI455X (gfx1250) — matrix operand order repaired and checked by reading, not compiled, not run on hardware
//
#include <hip/hip_runtime.h>
#include <hip/hip_bf16.h>

// ---------------------------------------------------------------------------
// BSpline-KAN fused forward for MI455X (gfx1250, wave32, WMMA).
//
// y = LN(silu(x)@bw.T + basis(x)@sw.T) -> affine -> silu, two layers.
// Fused as one K=2560 bf16 WMMA GEMM per layer with f32 accumulation.
// Weights stream through GLOBAL_LOAD_ASYNC_TO_LDS (ASYNCcnt) when available.
// ---------------------------------------------------------------------------

#define FEAT    512
#define KTOT    2560              // 512 (base) + 2048 (spline)
#define ASTRIDE 2568              // A row stride in LDS (+16B pad -> bank spread)
#define MTILE   32
#define NROWS   16384             // 4 * 4096
#define SEQ     4096
#define NKT     (KTOT / 32)       // 80 k-tiles

#define ABYTES  (MTILE * ASTRIDE * 2)          // 164352
#define BROWS   40                              // B row stride (ushorts): 80B, bank-spread
#define BBYTES  (8 * 2 * 64 * BROWS * 2)        // 8 waves * 2 bufs * 64 rows: 81920
#define REDBYTES ((32 * 8 * 2 + 64) * 4)        // 2304
#define SHBYTES (ABYTES + BBYTES + REDBYTES)    // 248576 <= 320KB WGP LDS

#if defined(__HIP_DEVICE_COMPILE__) && \
    __has_builtin(__builtin_amdgcn_global_load_async_to_lds_b128) && \
    __has_builtin(__builtin_amdgcn_s_wait_asynccnt)
#define USE_ASYNC 1
#endif

typedef __attribute__((ext_vector_type(16))) __bf16 v16bf;
typedef __attribute__((ext_vector_type(8)))  float  v8f;
typedef __attribute__((ext_vector_type(4)))  int    v4i;

#define AS1 __attribute__((address_space(1)))
#define AS3 __attribute__((address_space(3)))

static __device__ __forceinline__ unsigned short f2bf(float f) {
  unsigned int u = __float_as_uint(f);
  u += 0x7fffu + ((u >> 16) & 1u);         // round-to-nearest-even
  return (unsigned short)(u >> 16);
}

static __device__ __forceinline__ float sigmoidf_fast(float z) {
  return 1.0f / (1.0f + __expf(-z));
}

// First 4 cubic Cox-de Boor basis functions on knots {0,0,0,0,h,2h,3h,4h},
// h = 1/517 (matches reference KNOTS construction). Fully unrolled; knot
// arithmetic constant-folds to multiplies.
static __device__ __forceinline__ void bspline4(float x, float o[4]) {
  const float h = 1.0f / 517.0f;
  const float t[8] = {0.f, 0.f, 0.f, 0.f, h, 2.f * h, 3.f * h, 4.f * h};
  float B[7];
#pragma unroll
  for (int i = 0; i < 7; ++i)
    B[i] = (t[i] <= x && x < t[i + 1]) ? 1.0f : 0.0f;
#pragma unroll
  for (int d = 1; d <= 3; ++d) {
#pragma unroll
    for (int i = 0; i + d < 7; ++i) {
      float den1 = t[i + d] - t[i];
      float den2 = t[i + d + 1] - t[i + 1];
      float w1 = (den1 > 0.f) ? (x - t[i]) * (1.0f / den1) : 0.f;
      float w2 = (den2 > 0.f) ? (t[i + d + 1] - x) * (1.0f / den2) : 0.f;
      B[i] = w1 * B[i] + w2 * B[i + 1];
    }
  }
#pragma unroll
  for (int j = 0; j < 4; ++j) o[j] = B[j];
}

// --------------------------- weight packing -------------------------------
__global__ void pack_weights(const float* __restrict__ bw,
                             const float* __restrict__ sw,
                             unsigned short* __restrict__ wp) {
  int i = blockIdx.x * 256 + threadIdx.x;
  if (i >= FEAT * KTOT) return;
  int n = i / KTOT;
  int k = i - n * KTOT;
  float v = (k < FEAT) ? bw[n * FEAT + k] : sw[n * (4 * FEAT) + (k - FEAT)];
  wp[i] = f2bf(v);
}

// --------------------------- seq min/max ----------------------------------
__global__ void minmax_part(const float* __restrict__ act,
                            float* __restrict__ pmin,
                            float* __restrict__ pmax) {
  int id = blockIdx.x;                 // 256 = b(4) * sc(32) * fc(2)
  int fc = id & 1, sc = (id >> 1) & 31, b = id >> 6;
  int f = fc * 256 + threadIdx.x;
  float mn = 3.4e38f, mx = -3.4e38f;
  const float* p = act + ((size_t)b * SEQ + (size_t)sc * 128) * FEAT + f;
  for (int s = 0; s < 128; ++s) {
    float v = p[(size_t)s * FEAT];
    mn = fminf(mn, v);
    mx = fmaxf(mx, v);
  }
  pmin[((size_t)b * 32 + sc) * FEAT + f] = mn;
  pmax[((size_t)b * 32 + sc) * FEAT + f] = mx;
}

__global__ void minmax_red(const float* __restrict__ pmin,
                           const float* __restrict__ pmax,
                           float* __restrict__ xm, float* __restrict__ xM) {
  int id = blockIdx.x;                 // 8 = b(4) * fc(2)
  int fc = id & 1, b = id >> 1;
  int f = fc * 256 + threadIdx.x;
  float mn = 3.4e38f, mx = -3.4e38f;
  for (int c = 0; c < 32; ++c) {
    mn = fminf(mn, pmin[((size_t)b * 32 + c) * FEAT + f]);
    mx = fmaxf(mx, pmax[((size_t)b * 32 + c) * FEAT + f]);
  }
  xm[b * FEAT + f] = mn;
  xM[b * FEAT + f] = mx;
}

// --------------------------- fused KAN layer ------------------------------
// Block: 256 threads (8 wave32), 32 output rows, full N=512 per block so
// LayerNorm is block-local.
__global__ void __launch_bounds__(256)
kan_fused(const float* __restrict__ act, const unsigned short* __restrict__ wp,
          const float* __restrict__ xm, const float* __restrict__ xM,
          const float* __restrict__ gvec, const float* __restrict__ bvec,
          float* __restrict__ outp) {
  extern __shared__ char smem[];
  unsigned short* Abuf = (unsigned short*)smem;            // [MTILE][ASTRIDE]
  unsigned short* Bbuf = (unsigned short*)(smem + ABYTES); // per-wave dbl-buf slabs
  float* red1 = (float*)(smem + ABYTES + BBYTES);          // 32*8
  float* red2 = red1 + 32 * 8;                             // 32*8
  float* smu  = red2 + 32 * 8;                             // 32
  float* srs  = smu + 32;                                  // 32

  const int tid  = threadIdx.x;
  const int lane = tid & 31;
  const int w    = tid >> 5;
  const int half = lane >> 4;    // 0|1 (WMMA lane-half per ISA layouts)
  const int l15  = lane & 15;
  const int m0   = blockIdx.x * MTILE;
  const int b    = m0 >> 12;     // 4096 rows per batch, MTILE | 4096

  // ---- stage A = [silu(x) | bspline basis] as bf16 into LDS ----
  for (int i = tid; i < MTILE * FEAT; i += 256) {
    int r = i >> 9, f = i & (FEAT - 1);
    float xv = act[(size_t)(m0 + r) * FEAT + f];
    Abuf[r * ASTRIDE + f] = f2bf(xv * sigmoidf_fast(xv));
    float lo = xm[b * FEAT + f], hi = xM[b * FEAT + f];
    float xn = (xv - lo) / (hi - lo);
    float bs[4];
    bspline4(xn, bs);
    uint2 pk;
    pk.x = (unsigned)f2bf(bs[0]) | ((unsigned)f2bf(bs[1]) << 16);
    pk.y = (unsigned)f2bf(bs[2]) | ((unsigned)f2bf(bs[3]) << 16);
    *(uint2*)&Abuf[r * ASTRIDE + FEAT + 4 * f] = pk;   // K = 512 + 4f + j
  }
  __syncthreads();

  // ---- WMMA GEMM: [32 x 2560] x [2560 x 512], f32 accumulate ----
  const int nbase = w * 64;      // each wave owns 4 N-tiles of 16
  v8f acc[2][4];
#pragma unroll
  for (int mi = 0; mi < 2; ++mi)
#pragma unroll
    for (int ni = 0; ni < 4; ++ni) {
      v8f z = {0.f, 0.f, 0.f, 0.f, 0.f, 0.f, 0.f, 0.f};
      acc[mi][ni] = z;
    }

#if defined(USE_ASYNC)
  // Per-wave async weight staging: each wave double-buffers its own
  // [64 rows x 32 K] bf16 slab in LDS via GLOBAL_LOAD_ASYNC_TO_LDS_B128.
  // 8 chunks of 16B per lane per slab; per-wave ASYNCcnt orders completion.
  const unsigned wbase = (unsigned)w * (2u * 64u * BROWS);   // ushort index
  const unsigned short* gp[8];
  unsigned lo8[8];
#pragma unroll
  for (int c = 0; c < 8; ++c) {
    int idx = c * 32 + lane;           // 256 chunks: 64 rows x 4 x 16B
    int n_off = idx >> 2, ko = idx & 3;
    gp[c]  = wp + (size_t)(nbase + n_off) * KTOT + ko * 8;
    lo8[c] = wbase + (unsigned)n_off * BROWS + (unsigned)ko * 8;
  }
  auto stage_slab = [&](int nb) {
#pragma unroll
    for (int c = 0; c < 8; ++c) {
      v4i* gg = (v4i*)(size_t)gp[c];   // generic, const stripped via int round-trip
      v4i* ll = (v4i*)(Bbuf + (unsigned)nb * (64u * BROWS) + lo8[c]);
      __builtin_amdgcn_global_load_async_to_lds_b128(
          (AS1 v4i*)gg, (AS3 v4i*)ll, 0, 0);
      gp[c] += 32;                     // advance one k-slab
    }
  };
  stage_slab(0);
#endif

  for (int kt = 0; kt < NKT; ++kt) {
    const int k0 = kt * 32;
#if defined(USE_ASYNC)
    const int cur = kt & 1;
    if (kt + 1 < NKT) {
      stage_slab(cur ^ 1);
      __builtin_amdgcn_s_wait_asynccnt(8);   // slab kt complete (in-order)
    } else {
      __builtin_amdgcn_s_wait_asynccnt(0);
    }
    asm volatile("" ::: "memory");           // keep ds reads below the wait
#endif
    // A frags (ISA 16-bit A 16x32): lane half g holds K {k0+8g..+7, k0+16+8g..+7}
    v16bf afr[2];
#pragma unroll
    for (int mi = 0; mi < 2; ++mi) {
      const unsigned short* p = &Abuf[(mi * 16 + l15) * ASTRIDE + k0 + 8 * half];
      union { uint4 q[2]; v16bf v; } u;
      u.q[0] = *(const uint4*)p;             // ds_load_b128
      u.q[1] = *(const uint4*)(p + 16);
      afr[mi] = u.v;
    }
    // B frags: lane = col (l15), K {8*half..+7, 16+8*half..+7} within slab
    v16bf bfr[4];
#pragma unroll
    for (int ni = 0; ni < 4; ++ni) {
#if defined(USE_ASYNC)
      const unsigned short* p =
          &Bbuf[wbase + (unsigned)cur * (64u * BROWS) +
                (unsigned)(ni * 16 + l15) * BROWS + 8u * half];
#else
      const unsigned short* p =
          &wp[(size_t)(nbase + ni * 16 + l15) * KTOT + k0 + 8 * half];
#endif
      union { uint4 q[2]; v16bf v; } u;
      u.q[0] = *(const uint4*)p;
      u.q[1] = *(const uint4*)(p + 16);
      bfr[ni] = u.v;
    }
#pragma unroll
    for (int mi = 0; mi < 2; ++mi)
#pragma unroll
      for (int ni = 0; ni < 4; ++ni)
        acc[mi][ni] = __builtin_amdgcn_wmma_f32_16x16x32_bf16(
            false, afr[mi], false, bfr[ni], (short)0, acc[mi][ni], false, false);
  }

  // ---- LayerNorm stats: row r = mi*16 + 8*half + j, N=512 per row ----
  float s1[2][8], s2[2][8];
#pragma unroll
  for (int mi = 0; mi < 2; ++mi)
#pragma unroll
    for (int j = 0; j < 8; ++j) {
      float a = 0.f, q = 0.f;
#pragma unroll
      for (int ni = 0; ni < 4; ++ni) {
        float v = acc[mi][ni][j];
        a += v;
        q += v * v;
      }
      s1[mi][j] = a;
      s2[mi][j] = q;
    }
#pragma unroll
  for (int off = 1; off < 16; off <<= 1)     // reduce within 16-lane halves
#pragma unroll
    for (int mi = 0; mi < 2; ++mi)
#pragma unroll
      for (int j = 0; j < 8; ++j) {
        s1[mi][j] += __shfl_xor(s1[mi][j], off, 32);
        s2[mi][j] += __shfl_xor(s2[mi][j], off, 32);
      }
  if (l15 == 0) {
#pragma unroll
    for (int mi = 0; mi < 2; ++mi)
#pragma unroll
      for (int j = 0; j < 8; ++j) {
        int row = mi * 16 + 8 * half + j;
        red1[row * 8 + w] = s1[mi][j];
        red2[row * 8 + w] = s2[mi][j];
      }
  }
  __syncthreads();
  if (tid < 32) {
    float a = 0.f, q = 0.f;
#pragma unroll
    for (int ww = 0; ww < 8; ++ww) {
      a += red1[tid * 8 + ww];
      q += red2[tid * 8 + ww];
    }
    float mu = a * (1.0f / FEAT);
    float var = q * (1.0f / FEAT) - mu * mu;
    smu[tid] = mu;
    srs[tid] = rsqrtf(var + 1e-5f);
  }
  __syncthreads();

  // ---- epilogue: (y-mu)*rsig * g + b -> silu -> store f32 ----
#pragma unroll
  for (int ni = 0; ni < 4; ++ni) {
    int n = nbase + ni * 16 + l15;
    float gv = gvec[n], bv = bvec[n];
#pragma unroll
    for (int mi = 0; mi < 2; ++mi)
#pragma unroll
      for (int j = 0; j < 8; ++j) {
        int r = mi * 16 + 8 * half + j;
        float yn = (acc[mi][ni][j] - smu[r]) * srs[r];
        float z = yn * gv + bv;
        outp[(size_t)(m0 + r) * FEAT + n] = z * sigmoidf_fast(z);
      }
  }
}

// --------------------------------------------------------------------------
extern "C" void kernel_launch(void* const* d_in, const int* in_sizes, int n_in,
                              void* d_out, int out_size, void* d_ws, size_t ws_size,
                              hipStream_t stream) {
  (void)in_sizes; (void)n_in; (void)out_size; (void)ws_size;
  const float* x   = (const float*)d_in[0];
  const float* bw0 = (const float*)d_in[1];
  const float* sw0 = (const float*)d_in[2];
  const float* g0  = (const float*)d_in[3];
  const float* b0  = (const float*)d_in[4];
  const float* bw1 = (const float*)d_in[5];
  const float* sw1 = (const float*)d_in[6];
  const float* g1  = (const float*)d_in[7];
  const float* b1  = (const float*)d_in[8];
  float* out = (float*)d_out;

  char* ws = (char*)d_ws;
  const size_t H_BYTES  = (size_t)NROWS * FEAT * 4;   // 32 MB intermediate h
  const size_t WP_BYTES = (size_t)FEAT * KTOT * 2;    // 2.5 MB packed weights

  float*          h    = (float*)ws;
  unsigned short* wp0  = (unsigned short*)(ws + H_BYTES);
  unsigned short* wp1  = (unsigned short*)(ws + H_BYTES + WP_BYTES);
  float*          pmin = (float*)(ws + H_BYTES + 2 * WP_BYTES);
  float*          pmax = pmin + 4 * 32 * FEAT;
  float*          xm   = pmax + 4 * 32 * FEAT;
  float*          xM   = xm + 4 * FEAT;

  const int PACK_GRID = (FEAT * KTOT) / 256;          // 5120

  (void)hipFuncSetAttribute((const void*)kan_fused,
                            hipFuncAttributeMaxDynamicSharedMemorySize,
                            (int)SHBYTES);

  pack_weights<<<PACK_GRID, 256, 0, stream>>>(bw0, sw0, wp0);
  pack_weights<<<PACK_GRID, 256, 0, stream>>>(bw1, sw1, wp1);

  // layer 0: x -> h
  minmax_part<<<256, 256, 0, stream>>>(x, pmin, pmax);
  minmax_red<<<8, 256, 0, stream>>>(pmin, pmax, xm, xM);
  kan_fused<<<NROWS / MTILE, 256, SHBYTES, stream>>>(x, wp0, xm, xM, g0, b0, h);

  // layer 1: h -> out
  minmax_part<<<256, 256, 0, stream>>>(h, pmin, pmax);
  minmax_red<<<8, 256, 0, stream>>>(pmin, pmax, xm, xM);
  kan_fused<<<NROWS / MTILE, 256, SHBYTES, stream>>>(h, wp1, xm, xM, g1, b1, out);
}
